// SimpleGraphAttention_12962211299598
// MI455X (gfx1250) — compile-verified, matrix operand order checked by reading, not run on hardware
//
#include <hip/hip_runtime.h>

// ---------------------------------------------------------------------------
// Types for CDNA5 WMMA (wave32): v_wmma_f32_16x16x32_bf16
// ---------------------------------------------------------------------------
typedef __bf16 bf16_t;
typedef bf16_t bf16x8 __attribute__((ext_vector_type(8)));   // 16 bytes
typedef bf16_t v16bf  __attribute__((ext_vector_type(16)));  // A/B fragment
typedef float  v8f    __attribute__((ext_vector_type(8)));   // C/D fragment

#define NNODES 8192
#define INDIM  256
#define HIDDEN 512
#define BN     128      // key block for fused attention

// ---------------------------------------------------------------------------
// Fragment load, 16-bit A-matrix 16x32 layout (ISA 7.12.2): lane L(0..15) is
// row M=L; holds K = hi*8+[0..7] and 16+hi*8+[0..7] (hi = lane>>4) as two
// contiguous 16-byte chunks. The B operand (K x N) uses the identical per-lane
// pattern indexed by N, so a row-major N x K matrix loads as a valid B frag.
// ---------------------------------------------------------------------------
__device__ __forceinline__ v16bf load_frag(const bf16_t* __restrict__ X,
                                           int ld, int row, int k0, int lane) {
  const int lrow = lane & 15;
  const int hi   = (lane >> 4) & 1;
  const bf16_t* p = X + (size_t)(row + lrow) * (size_t)ld + (size_t)(k0 + hi * 8);
  bf16x8 a = *(const bf16x8*)(p);
  bf16x8 b = *(const bf16x8*)(p + 16);
  v16bf r;
#pragma unroll
  for (int i = 0; i < 8; ++i) { r[i] = a[i]; r[i + 8] = b[i]; }
  return r;
}

template <int LD>
__device__ __forceinline__ v16bf load_frag_lds(const bf16_t (*buf)[LD],
                                               int row, int k0, int lane) {
  const int lrow = lane & 15;
  const int hi   = (lane >> 4) & 1;
  const bf16_t* p = &buf[row + lrow][k0 + hi * 8];
  bf16x8 a = *(const bf16x8*)(p);
  bf16x8 b = *(const bf16x8*)(p + 16);
  v16bf r;
#pragma unroll
  for (int i = 0; i < 8; ++i) { r[i] = a[i]; r[i + 8] = b[i]; }
  return r;
}

// ---------------------------------------------------------------------------
// Generic bf16 WMMA GEMM:  C[M,N] = A[M,K] @ Bt[N,K]^T  (+ epilogue)
// One wave -> 64x32 output tile (4x2 fragments). Block = 8 waves.
// ---------------------------------------------------------------------------
__global__ void __launch_bounds__(256)
gemm_bf16_wmma(const bf16_t* __restrict__ A, const bf16_t* __restrict__ Bt,
               const float* __restrict__ bias, const float* __restrict__ residual,
               int relu_flag,
               float* __restrict__ outF, bf16_t* __restrict__ outB,
               int M, int N, int K)
{
  const int lane = threadIdx.x & 31;
  const int wave = (int)((blockIdx.x * blockDim.x + threadIdx.x) >> 5);
  const int ntn  = N >> 5;
  const int mt   = wave / ntn;
  const int nt   = wave - mt * ntn;
  const int m0   = mt * 64;
  const int n0   = nt * 32;
  if (m0 >= M) return;   // wave-uniform

  v8f acc[4][2];
#pragma unroll
  for (int i = 0; i < 4; ++i)
#pragma unroll
    for (int j = 0; j < 2; ++j)
      acc[i][j] = (v8f){0.f, 0.f, 0.f, 0.f, 0.f, 0.f, 0.f, 0.f};

  for (int k0 = 0; k0 < K; k0 += 32) {
    v16bf a[4];
#pragma unroll
    for (int i = 0; i < 4; ++i) a[i] = load_frag(A, K, m0 + 16 * i, k0, lane);
    v16bf b[2];
#pragma unroll
    for (int j = 0; j < 2; ++j) b[j] = load_frag(Bt, K, n0 + 16 * j, k0, lane);
#pragma unroll
    for (int i = 0; i < 4; ++i)
#pragma unroll
      for (int j = 0; j < 2; ++j)
        acc[i][j] = __builtin_amdgcn_wmma_f32_16x16x32_bf16(
            false, a[i], false, b[j], (short)0, acc[i][j], false, false);
  }

  const int l15 = lane & 15;
  const int hi  = (lane >> 4) & 1;
#pragma unroll
  for (int j = 0; j < 2; ++j) {
    const int cn = n0 + 16 * j + l15;
    const float bv = bias ? bias[cn] : 0.0f;
#pragma unroll
    for (int i = 0; i < 4; ++i) {
      const int rm = m0 + 16 * i + hi * 8;
#pragma unroll
      for (int r = 0; r < 8; ++r) {
        const int cm = rm + r;
        const size_t idx = (size_t)cm * (size_t)N + (size_t)cn;
        float v = acc[i][j][r] + bv;
        if (relu_flag) v = v > 0.0f ? v : 0.0f;
        if (residual)  v += residual[idx];
        if (outF) outF[idx] = v;
        if (outB) outB[idx] = (bf16_t)v;
      }
    }
  }
}

// ---------------------------------------------------------------------------
// Fused masked-softmax attention (flash-style):
//   h2 = softmax(mask(q @ k^T * scale)) @ v
// Block: 512 threads = 16 waves = 4 row-groups x 4 col-waves; 64 q rows/block.
// Per wave: 16 persistent q A-frags (full K=512) + 8 accum frags (16x128 out).
// Key blocks of BN=128: WMMA scores -> LDS -> masked online softmax (thread
// stage, vectorized adj read) -> bf16 p in LDS -> rescale accumulators ->
// WMMA p@v (A from LDS, B from L2-resident vT).
// ---------------------------------------------------------------------------
__global__ void __launch_bounds__(512)
flash_attn_kernel(const bf16_t* __restrict__ q, const bf16_t* __restrict__ k,
                  const bf16_t* __restrict__ vT,
                  const unsigned char* __restrict__ adj,
                  bf16_t* __restrict__ h2, float scale)
{
  __shared__ float  s_buf[64][BN];   // 32KB raw scores
  __shared__ bf16_t p_buf[64][BN];   // 16KB softmaxed probabilities
  __shared__ float  m_arr[64];       // running row max
  __shared__ float  l_arr[64];       // running row sum
  __shared__ float  rs_arr[64];      // per-block rescale factor
  __shared__ float  red[64][8];      // per-row partial reductions

  const int tid   = threadIdx.x;
  const int lane  = tid & 31;
  const int wv    = tid >> 5;          // 0..15
  const int rg    = wv >> 2;           // row group 0..3 (16 rows each)
  const int cw    = wv & 3;            // col wave 0..3 (128 out cols each)
  const int qbase = blockIdx.x * 64;
  const int l15   = lane & 15;
  const int hi8   = ((lane >> 4) & 1) * 8;

  if (tid < 64) { m_arr[tid] = -1.0e30f; l_arr[tid] = 0.0f; }

  // persistent q fragments: 16 rows x K=512 -> 16 frags (128 VGPRs)
  v16bf qa[16];
#pragma unroll
  for (int kf = 0; kf < 16; ++kf)
    qa[kf] = load_frag(q, HIDDEN, qbase + rg * 16, kf * 32, lane);

  v8f acc[8];
#pragma unroll
  for (int f = 0; f < 8; ++f)
    acc[f] = (v8f){0.f, 0.f, 0.f, 0.f, 0.f, 0.f, 0.f, 0.f};

  const int srow = tid >> 3;           // 0..63 (softmax stage: 8 thr/row)
  const int sub  = tid & 7;
  const int c0   = sub * 16;

  __syncthreads();

  for (int kb = 0; kb < NNODES; kb += BN) {
    // ---- stage 1: score slice [16 rows x 32 cols], K = 512 ---------------
    // Two half-passes (one 16-col tile each) to keep peak VGPR liveness low:
    // only one B fragment + one score accumulator live beside qa/acc.
#pragma unroll
    for (int j = 0; j < 2; ++j) {
      v8f s = (v8f){0.f, 0.f, 0.f, 0.f, 0.f, 0.f, 0.f, 0.f};
#pragma unroll
      for (int kf = 0; kf < 16; ++kf) {
        v16bf b = load_frag(k, HIDDEN, kb + cw * 32 + 16 * j, kf * 32, lane);
        s = __builtin_amdgcn_wmma_f32_16x16x32_bf16(false, qa[kf], false, b,
                                                    (short)0, s, false, false);
      }
#pragma unroll
      for (int r = 0; r < 8; ++r)
        s_buf[rg * 16 + hi8 + r][cw * 32 + 16 * j + l15] = s[r];
    }
    __syncthreads();

    // ---- stage 2: masked online softmax over the 128-wide block ----------
    // adj bytes fetched as a single 16B vector (always 16B aligned).
    const unsigned char* arow =
        adj + (size_t)(qbase + srow) * (size_t)NNODES + (size_t)(kb + c0);
    const uint4 am = *(const uint4*)arow;
    unsigned int aw[4] = {am.x, am.y, am.z, am.w};
    float vv[16];
    float lmax = -1.0e30f;
#pragma unroll
    for (int i = 0; i < 16; ++i) {
      float sv = s_buf[srow][c0 + i] * scale;
      const unsigned int abyte = (aw[i >> 2] >> ((i & 3) * 8)) & 0xffu;
      sv = abyte ? sv : -1.0e30f;       // mask value matches the reference
      vv[i] = sv;
      lmax = fmaxf(lmax, sv);
    }
    red[srow][sub] = lmax;
    __syncthreads();
    if (sub == 0) {
      float mx = red[srow][0];
#pragma unroll
      for (int i = 1; i < 8; ++i) mx = fmaxf(mx, red[srow][i]);
      const float mo = m_arr[srow];
      const float mn = fmaxf(mo, mx);
      m_arr[srow]  = mn;
      rs_arr[srow] = __expf(mo - mn);   // rescale for old acc / old l
    }
    __syncthreads();
    const float mn = m_arr[srow];
    float lsum = 0.0f;
#pragma unroll
    for (int i = 0; i < 16; ++i) {
      const float p = __expf(vv[i] - mn);
      p_buf[srow][c0 + i] = (bf16_t)p;
      lsum += p;
    }
    red[srow][sub] = lsum;
    __syncthreads();
    if (sub == 0) {
      float sm = 0.0f;
#pragma unroll
      for (int i = 0; i < 8; ++i) sm += red[srow][i];
      l_arr[srow] = l_arr[srow] * rs_arr[srow] + sm;
    }
    __syncthreads();

    // ---- stage 3: rescale accumulators, accumulate p @ v -----------------
    {
      float sc[8];
#pragma unroll
      for (int r = 0; r < 8; ++r) sc[r] = rs_arr[rg * 16 + hi8 + r];
#pragma unroll
      for (int f = 0; f < 8; ++f)
#pragma unroll
        for (int r = 0; r < 8; ++r) acc[f][r] *= sc[r];
    }
#pragma unroll
    for (int k0 = 0; k0 < BN; k0 += 32) {
      v16bf a = load_frag_lds<BN>(p_buf, rg * 16, k0, lane);
#pragma unroll
      for (int f = 0; f < 8; ++f) {
        v16bf b = load_frag(vT, NNODES, cw * 128 + 16 * f, kb + k0, lane);
        acc[f] = __builtin_amdgcn_wmma_f32_16x16x32_bf16(
            false, a, false, b, (short)0, acc[f], false, false);
      }
    }
    __syncthreads();   // p_buf fully consumed before next block rewrites it
  }

  // ---- epilogue: divide by l, store bf16 h2 ------------------------------
  float inv[8];
#pragma unroll
  for (int r = 0; r < 8; ++r) inv[r] = 1.0f / l_arr[rg * 16 + hi8 + r];
#pragma unroll
  for (int f = 0; f < 8; ++f) {
    const int col = cw * 128 + 16 * f + l15;
#pragma unroll
    for (int r = 0; r < 8; ++r) {
      const int row = qbase + rg * 16 + hi8 + r;
      h2[(size_t)row * HIDDEN + col] = (bf16_t)(acc[f][r] * inv[r]);
    }
  }
}

// ---------------------------------------------------------------------------
// LayerNorm over HID=512 per row. One block (256 thr) per row.
// ---------------------------------------------------------------------------
__global__ void __launch_bounds__(256)
layernorm_rows(const float* __restrict__ in, const float* __restrict__ g,
               const float* __restrict__ be, float* __restrict__ outF,
               bf16_t* __restrict__ outB)
{
  __shared__ float red[256];
  const int row = blockIdx.x;
  const int tid = threadIdx.x;
  const float* p = in + (size_t)row * HIDDEN;
  const float v0 = p[tid], v1 = p[tid + 256];
  red[tid] = v0 + v1; __syncthreads();
  for (int st = 128; st > 0; st >>= 1) {
    if (tid < st) red[tid] += red[tid + st];
    __syncthreads();
  }
  const float mean = red[0] * (1.0f / (float)HIDDEN);
  __syncthreads();
  const float d0 = v0 - mean, d1 = v1 - mean;
  red[tid] = d0 * d0 + d1 * d1; __syncthreads();
  for (int st = 128; st > 0; st >>= 1) {
    if (tid < st) red[tid] += red[tid + st];
    __syncthreads();
  }
  const float inv = rsqrtf(red[0] * (1.0f / (float)HIDDEN) + 1e-5f);
  const float o0 = d0 * inv * g[tid] + be[tid];
  const float o1 = d1 * inv * g[tid + 256] + be[tid + 256];
  float* qo = outF + (size_t)row * HIDDEN;
  qo[tid] = o0; qo[tid + 256] = o1;
  if (outB) {
    bf16_t* qb = outB + (size_t)row * HIDDEN;
    qb[tid] = (bf16_t)o0; qb[tid + 256] = (bf16_t)o1;
  }
}

// ---------------------------------------------------------------------------
// logits[row] = dot(h[row,:], Ws) + bs
// ---------------------------------------------------------------------------
__global__ void __launch_bounds__(256)
logits_kernel(const float* __restrict__ h, const float* __restrict__ Ws,
              const float* __restrict__ bs, float* __restrict__ out)
{
  __shared__ float red[256];
  const int row = blockIdx.x;
  const int tid = threadIdx.x;
  const float* p = h + (size_t)row * HIDDEN;
  red[tid] = p[tid] * Ws[tid] + p[tid + 256] * Ws[tid + 256];
  __syncthreads();
  for (int st = 128; st > 0; st >>= 1) {
    if (tid < st) red[tid] += red[tid + st];
    __syncthreads();
  }
  if (tid == 0) out[row] = red[0] + bs[0];
}

// ---------------------------------------------------------------------------
// Conversions / transposes.
// ---------------------------------------------------------------------------
__global__ void __launch_bounds__(256)
f32_to_bf16_kernel(const float* __restrict__ in, bf16_t* __restrict__ out, int n)
{
  const int i = blockIdx.x * blockDim.x + threadIdx.x;
  if (i < n) out[i] = (bf16_t)in[i];
}

__global__ void __launch_bounds__(256)
transpose_f32_to_bf16(const float* __restrict__ in, bf16_t* __restrict__ out,
                      int R, int C)
{
  const int i = blockIdx.x * blockDim.x + threadIdx.x;
  if (i >= R * C) return;
  const int r = i / C, c = i - r * C;
  out[(size_t)c * R + r] = (bf16_t)in[i];
}

__global__ void __launch_bounds__(256)
transpose_bf16(const bf16_t* __restrict__ in, bf16_t* __restrict__ out,
               int R, int C)
{
  const int i = blockIdx.x * blockDim.x + threadIdx.x;
  if (i >= R * C) return;
  const int r = i / C, c = i - r * C;
  out[(size_t)c * R + r] = in[i];
}

// ---------------------------------------------------------------------------
// Host orchestration
// ---------------------------------------------------------------------------
extern "C" void kernel_launch(void* const* d_in, const int* in_sizes, int n_in,
                              void* d_out, int out_size, void* d_ws, size_t ws_size,
                              hipStream_t stream)
{
  (void)in_sizes; (void)n_in; (void)out_size; (void)ws_size;
  const float*         x    = (const float*)d_in[0];
  const unsigned char* adj  = (const unsigned char*)d_in[1];  // jnp bool -> 1 byte
  const float* Win = (const float*)d_in[2];
  const float* b_in= (const float*)d_in[3];
  const float* Wq  = (const float*)d_in[4];
  const float* bq  = (const float*)d_in[5];
  const float* Wk  = (const float*)d_in[6];
  const float* bk  = (const float*)d_in[7];
  const float* Wv  = (const float*)d_in[8];
  const float* bv  = (const float*)d_in[9];
  const float* Wo  = (const float*)d_in[10];
  const float* bo  = (const float*)d_in[11];
  const float* Ws  = (const float*)d_in[12];
  const float* bs  = (const float*)d_in[13];
  const float* g1  = (const float*)d_in[14];
  const float* be1 = (const float*)d_in[15];
  const float* g2  = (const float*)d_in[16];
  const float* be2 = (const float*)d_in[17];
  const float* Wf1 = (const float*)d_in[18];
  const float* bf1 = (const float*)d_in[19];
  const float* Wf2 = (const float*)d_in[20];
  const float* bf2 = (const float*)d_in[21];
  float* out = (float*)d_out;

  char* ws = (char*)d_ws;
  size_t off = 0;
  auto alloc = [&](size_t bytes) -> void* {
    void* p = ws + off;
    off += (bytes + 255) & ~(size_t)255;
    return p;
  };

  const size_t NH = (size_t)NNODES * HIDDEN;
  bf16_t* xb     = (bf16_t*)alloc((size_t)NNODES * INDIM * 2);
  bf16_t* WinT   = (bf16_t*)alloc((size_t)HIDDEN * INDIM * 2);
  bf16_t* WqT    = (bf16_t*)alloc((size_t)HIDDEN * HIDDEN * 2);
  bf16_t* WkT    = (bf16_t*)alloc((size_t)HIDDEN * HIDDEN * 2);
  bf16_t* WvT    = (bf16_t*)alloc((size_t)HIDDEN * HIDDEN * 2);
  bf16_t* WoT    = (bf16_t*)alloc((size_t)HIDDEN * HIDDEN * 2);
  bf16_t* Wf1T   = (bf16_t*)alloc((size_t)HIDDEN * HIDDEN * 2);
  bf16_t* Wf2T   = (bf16_t*)alloc((size_t)HIDDEN * HIDDEN * 2);
  float*  h_f    = (float*)alloc(NH * 4);
  bf16_t* h_b    = (bf16_t*)alloc(NH * 2);
  bf16_t* q_b    = (bf16_t*)alloc(NH * 2);
  bf16_t* k_b    = (bf16_t*)alloc(NH * 2);
  bf16_t* v_b    = (bf16_t*)alloc(NH * 2);
  bf16_t* vT_b   = (bf16_t*)alloc(NH * 2);
  bf16_t* h2_b   = (bf16_t*)alloc(NH * 2);
  float*  pre1   = (float*)alloc(NH * 4);
  float*  hln1_f = (float*)alloc(NH * 4);
  bf16_t* hln1_b = (bf16_t*)alloc(NH * 2);
  bf16_t* ff1_b  = (bf16_t*)alloc(NH * 2);
  float*  pre2   = (float*)alloc(NH * 4);
  float*  hfin_f = (float*)alloc(NH * 4);

  const float scale = 0.044194173824159216f;  // 1/sqrt(512)

  // --- precision conversion / weight transposition -------------------------
  {
    const int n = NNODES * INDIM;
    f32_to_bf16_kernel<<<n / 256, 256, 0, stream>>>(x, xb, n);
  }
  transpose_f32_to_bf16<<<(INDIM * HIDDEN) / 256, 256, 0, stream>>>(Win, WinT, INDIM, HIDDEN);
  transpose_f32_to_bf16<<<(HIDDEN * HIDDEN) / 256, 256, 0, stream>>>(Wq,  WqT,  HIDDEN, HIDDEN);
  transpose_f32_to_bf16<<<(HIDDEN * HIDDEN) / 256, 256, 0, stream>>>(Wk,  WkT,  HIDDEN, HIDDEN);
  transpose_f32_to_bf16<<<(HIDDEN * HIDDEN) / 256, 256, 0, stream>>>(Wv,  WvT,  HIDDEN, HIDDEN);
  transpose_f32_to_bf16<<<(HIDDEN * HIDDEN) / 256, 256, 0, stream>>>(Wo,  WoT,  HIDDEN, HIDDEN);
  transpose_f32_to_bf16<<<(HIDDEN * HIDDEN) / 256, 256, 0, stream>>>(Wf1, Wf1T, HIDDEN, HIDDEN);
  transpose_f32_to_bf16<<<(HIDDEN * HIDDEN) / 256, 256, 0, stream>>>(Wf2, Wf2T, HIDDEN, HIDDEN);

  auto gemm_grid = [](int M, int N) { return (M / 64) * (N / 32) / 8; };

  // --- h = x@Win + b_in ----------------------------------------------------
  gemm_bf16_wmma<<<gemm_grid(NNODES, HIDDEN), 256, 0, stream>>>(
      xb, WinT, b_in, nullptr, 0, h_f, h_b, NNODES, HIDDEN, INDIM);

  // --- q, k, v -------------------------------------------------------------
  gemm_bf16_wmma<<<gemm_grid(NNODES, HIDDEN), 256, 0, stream>>>(
      h_b, WqT, bq, nullptr, 0, nullptr, q_b, NNODES, HIDDEN, HIDDEN);
  gemm_bf16_wmma<<<gemm_grid(NNODES, HIDDEN), 256, 0, stream>>>(
      h_b, WkT, bk, nullptr, 0, nullptr, k_b, NNODES, HIDDEN, HIDDEN);
  gemm_bf16_wmma<<<gemm_grid(NNODES, HIDDEN), 256, 0, stream>>>(
      h_b, WvT, bv, nullptr, 0, nullptr, v_b, NNODES, HIDDEN, HIDDEN);

  // vT for the p@v B operand
  transpose_bf16<<<(int)(NH / 256), 256, 0, stream>>>(v_b, vT_b, NNODES, HIDDEN);

  // --- fused attention: h2 = softmax(mask(q k^T / sqrt(d))) @ v ------------
  flash_attn_kernel<<<NNODES / 64, 512, 0, stream>>>(
      q_b, k_b, vT_b, adj, h2_b, scale);

  // --- pre1 = h + h2@Wo + bo ; h = LN1(pre1) -------------------------------
  gemm_bf16_wmma<<<gemm_grid(NNODES, HIDDEN), 256, 0, stream>>>(
      h2_b, WoT, bo, h_f, 0, pre1, nullptr, NNODES, HIDDEN, HIDDEN);
  layernorm_rows<<<NNODES, 256, 0, stream>>>(pre1, g1, be1, hln1_f, hln1_b);

  // --- ff = relu(h@Wf1+bf1)@Wf2 + bf2 ; h = LN2(h + ff) --------------------
  gemm_bf16_wmma<<<gemm_grid(NNODES, HIDDEN), 256, 0, stream>>>(
      hln1_b, Wf1T, bf1, nullptr, 1, nullptr, ff1_b, NNODES, HIDDEN, HIDDEN);
  gemm_bf16_wmma<<<gemm_grid(NNODES, HIDDEN), 256, 0, stream>>>(
      ff1_b, Wf2T, bf2, hln1_f, 0, pre2, nullptr, NNODES, HIDDEN, HIDDEN);
  layernorm_rows<<<NNODES, 256, 0, stream>>>(pre2, g2, be2, hfin_f, nullptr);

  // --- logits --------------------------------------------------------------
  logits_kernel<<<NNODES, 256, 0, stream>>>(hfin_f, Ws, bs, out);
}
